// NonLocalBlock_81965155877217
// MI455X (gfx1250) — matrix operand order repaired and checked by reading, not compiled, not run on hardware
//
#include <hip/hip_runtime.h>
#include <hip/hip_bf16.h>
#include <stdint.h>

typedef __bf16 v16bf __attribute__((ext_vector_type(16)));
typedef __bf16 v8bf  __attribute__((ext_vector_type(8)));
typedef float  v8f   __attribute__((ext_vector_type(8)));

constexpr int kB   = 8;
constexpr int kC1  = 256;
constexpr int kHW  = 3136;   // 56*56, divisible by 64
constexpr int kCI  = 32;
constexpr int kCG  = 128;
constexpr int kTP  = 64;     // pixels per tile
constexpr int kNT  = kHW / kTP;  // 49 tiles
constexpr float kRS = 0.99999500003749969f; // 1/sqrt(1+1e-5)

// ---------------------------------------------------------------------------
// CDNA5 async global->LDS copies (ASYNCcnt path), with sync fallback.
// Builtin signature (from hipcc diagnostic): params are pointers to
// 'int __attribute__((vector_size(16)))' in AS(1) / AS(3).
// ---------------------------------------------------------------------------
#if __has_builtin(__builtin_amdgcn_global_load_async_to_lds_b128) && \
    __has_builtin(__builtin_amdgcn_s_wait_asynccnt)
#define USE_ASYNC_LDS 1
typedef int async_v4i __attribute__((vector_size(16)));
typedef __attribute__((address_space(1))) async_v4i* async_gp;
typedef __attribute__((address_space(3))) async_v4i* async_lp;
#else
#define USE_ASYNC_LDS 0
#endif

static __device__ __forceinline__ void copy16_g2l(void* lds, const void* g) {
#if USE_ASYNC_LDS
    __builtin_amdgcn_global_load_async_to_lds_b128((async_gp)g, (async_lp)lds, 0, 0);
#else
    *(uint4*)lds = *(const uint4*)g;
#endif
}
static __device__ __forceinline__ void async_copy_wait() {
#if USE_ASYNC_LDS
    __builtin_amdgcn_s_wait_asynccnt(0);
#endif
}

// ---------------------------------------------------------------------------
// DPP 16-lane row reductions (keeps lanes 0-15 / 16-31 independent, matching
// the WMMA C-layout half-wave row split). row_ror:N = 0x120|N.
// ---------------------------------------------------------------------------
template <int CTRL>
static __device__ __forceinline__ float dpp_mov(float x) {
    return __builtin_bit_cast(float,
        __builtin_amdgcn_update_dpp(0, __builtin_bit_cast(int, x),
                                    CTRL, 0xF, 0xF, true));
}
static __device__ __forceinline__ float red16_max(float x) {
    x = fmaxf(x, dpp_mov<0x128>(x));   // row_ror:8
    x = fmaxf(x, dpp_mov<0x124>(x));   // row_ror:4
    x = fmaxf(x, dpp_mov<0x122>(x));   // row_ror:2
    x = fmaxf(x, dpp_mov<0x121>(x));   // row_ror:1
    return x;
}
static __device__ __forceinline__ float red16_sum(float x) {
    x += dpp_mov<0x128>(x);
    x += dpp_mov<0x124>(x);
    x += dpp_mov<0x122>(x);
    x += dpp_mov<0x121>(x);
    return x;
}

static __device__ __forceinline__ float silu(float y) {
    return y / (1.0f + __expf(-y));
}

// Build a v16bf from two 16-byte chunks (each 16B-aligned).
static __device__ __forceinline__ v16bf ld2(const __bf16* p0, const __bf16* p1) {
    v16bf r;
    *((v8bf*)&r)       = *(const v8bf*)p0;
    *(((v8bf*)&r) + 1) = *(const v8bf*)p1;
    return r;
}

// ---------------------------------------------------------------------------
// Kernel 1: fused theta/phi/g 1x1-conv + BN + SiLU, emit bf16.
// theta, phi -> pixel-major [b][p][32]; g -> channel-major [b][128][p].
// Block: 256 threads = 64 pixels x 4 groups of 48 output channels.
// ---------------------------------------------------------------------------
__global__ __launch_bounds__(256) void proj_kernel(
    const float* __restrict__ x,
    const float* __restrict__ w_th, const float* __restrict__ w_ph,
    const float* __restrict__ w_g,
    const float* __restrict__ g_th, const float* __restrict__ b_th,
    const float* __restrict__ g_ph, const float* __restrict__ b_ph,
    const float* __restrict__ g_g,  const float* __restrict__ b_g,
    __bf16* __restrict__ thW, __bf16* __restrict__ phW, __bf16* __restrict__ gW)
{
    __shared__ __align__(16) float ldsX[32][68];   // 32 channels x 64 px (+pad)
    const int blk = blockIdx.x;
    const int b = blk / kNT, pt = blk % kNT;
    const int p0 = pt * kTP;
    const int t = threadIdx.x;
    const int pix = t & 63, grp = t >> 6;

    float acc[48];
#pragma unroll
    for (int j = 0; j < 48; ++j) acc[j] = 0.0f;

    for (int cc = 0; cc < kC1; cc += 32) {
        __syncthreads();
#pragma unroll
        for (int i = 0; i < 2; ++i) {
            int idx = t + i * 256;            // 512 uint4 chunks total
            int r = idx >> 4, ch = idx & 15;  // row in chunk, 16B chunk in row
            copy16_g2l(&ldsX[r][ch * 4],
                       x + (size_t)(b * kC1 + cc + r) * kHW + p0 + ch * 4);
        }
        async_copy_wait();
        __syncthreads();
#pragma unroll 4
        for (int r = 0; r < 32; ++r) {
            float xv = ldsX[r][pix];
            int c = cc + r;
#pragma unroll
            for (int j = 0; j < 48; ++j) {
                int o = grp * 48 + j;   // uniform per wave
                float wv;
                if (o < 32)      wv = w_th[o * kC1 + c];
                else if (o < 64) wv = w_ph[(o - 32) * kC1 + c];
                else             wv = w_g[(o - 64) * kC1 + c];
                acc[j] = fmaf(xv, wv, acc[j]);
            }
        }
    }

    const size_t prow = (size_t)b * kHW + p0 + pix;
#pragma unroll
    for (int j = 0; j < 48; ++j) {
        int o = grp * 48 + j;
        float gam, bet;
        if (o < 32)      { gam = g_th[o];      bet = b_th[o];      }
        else if (o < 64) { gam = g_ph[o - 32]; bet = b_ph[o - 32]; }
        else             { gam = g_g[o - 64];  bet = b_g[o - 64];  }
        float y = silu(acc[j] * gam * kRS + bet);
        if (o < 32)      thW[prow * kCI + o]        = (__bf16)y;
        else if (o < 64) phW[prow * kCI + (o - 32)] = (__bf16)y;
        else             gW[((size_t)(b * kCG + (o - 64))) * kHW + p0 + pix] = (__bf16)y;
    }
}

// ---------------------------------------------------------------------------
// Kernel 2: flash-attention style fused softmax(theta @ phi^T) @ G using
// v_wmma_f32_16x16x32_bf16. 128 threads = 4 waves; wave w owns 16 query rows.
// ---------------------------------------------------------------------------
__global__ __launch_bounds__(128) void attn_kernel(
    const __bf16* __restrict__ thW, const __bf16* __restrict__ phW,
    const __bf16* __restrict__ gW,  __bf16* __restrict__ OW)
{
    __shared__ __align__(16) __bf16 sPhi[64 * 40];       // [m][c] rows of 40 halfs
    __shared__ __align__(16) __bf16 sG[128 * 72];        // [ch][m] rows of 72 halfs
    __shared__ __align__(16) __bf16 sP[4 * 16 * 72];     // per-wave P, [row][m]

    const int blk = blockIdx.x;
    const int b = blk / kNT, nt = blk % kNT;
    const int t = threadIdx.x;
    const int lane = t & 31, wid = t >> 5;
    const int l15 = lane & 15;
    const bool hi = lane >= 16;
    const int sel = hi ? 8 : 0;
    const int rowbase = nt * kTP + wid * 16;

    // Theta A operand (16 rows x K=32), ISA 16-bit 16x32 A layout.
    const __bf16* trow = thW + ((size_t)b * kHW + rowbase + l15) * kCI;
    const v16bf Ath = ld2(trow + sel, trow + 16 + sel);

    v8f acc[8];
    v8f zero = {};
#pragma unroll
    for (int ct = 0; ct < 8; ++ct) acc[ct] = zero;
    float mrow[8], lrow[8];
#pragma unroll
    for (int v = 0; v < 8; ++v) { mrow[v] = -1e30f; lrow[v] = 0.0f; }

    __bf16* pbase = sP + wid * 16 * 72;

    for (int mt = 0; mt < kNT; ++mt) {
        const int m0 = mt * kTP;
        __syncthreads();   // previous iteration done with sPhi/sG
        {   // phi tile: 64 rows x 32 halfs (two 16B chunks per thread)
            int r = t >> 1, hb = (t & 1) * 16;
            const __bf16* src = phW + ((size_t)b * kHW + m0 + r) * kCI + hb;
            __bf16* dst = sPhi + r * 40 + hb;
            copy16_g2l(dst, src);
            copy16_g2l(dst + 8, src + 8);
        }
        {   // G tile: channel-major, 128 rows x 64 halfs (8 x 16B per thread)
            const __bf16* src = gW + ((size_t)(b * kCG + t)) * kHW + m0;
            __bf16* dst = sG + t * 72;
#pragma unroll
            for (int i = 0; i < 8; ++i) copy16_g2l(dst + i * 8, src + i * 8);
        }
        async_copy_wait();
        __syncthreads();

        // S tile: 16 rows x 64 cols via 4 WMMAs (B = phi^T in B-operand layout)
        v8f S[4];
#pragma unroll
        for (int tt = 0; tt < 4; ++tt) {
            const __bf16* pr = sPhi + (tt * 16 + l15) * 40 + (hi ? 8 : 0);
            v16bf Bp = ld2(pr, pr + 16);
            v8f z = {};
            S[tt] = __builtin_amdgcn_wmma_f32_16x16x32_bf16(
                false, Ath, false, Bp, (short)0, z, false, false);
        }

        // online softmax: row max via DPP row reductions (no LDS traffic)
        float scale[8];
#pragma unroll
        for (int v = 0; v < 8; ++v) {
            float m = fmaxf(fmaxf(S[0][v], S[1][v]), fmaxf(S[2][v], S[3][v]));
            m = red16_max(m);
            float mn = fmaxf(mrow[v], m);
            scale[v] = __expf(mrow[v] - mn);
            mrow[v] = mn;
        }

        // P = exp(S - m): accumulate row sums + stage to LDS (C -> A transpose)
        float rsum[8];
#pragma unroll
        for (int v = 0; v < 8; ++v) rsum[v] = 0.0f;
#pragma unroll
        for (int tt = 0; tt < 4; ++tt) {
#pragma unroll
            for (int v = 0; v < 8; ++v) {
                float pv = __expf(S[tt][v] - mrow[v]);
                rsum[v] += pv;
                pbase[(v + (hi ? 8 : 0)) * 72 + tt * 16 + l15] = (__bf16)pv;
            }
        }
#pragma unroll
        for (int v = 0; v < 8; ++v)
            lrow[v] = lrow[v] * scale[v] + red16_sum(rsum[v]);
#pragma unroll
        for (int ct = 0; ct < 8; ++ct)
#pragma unroll
            for (int v = 0; v < 8; ++v) acc[ct][v] *= scale[v];

        __syncthreads();   // P staged (and orders sG reads below vs. next fill)

        // O += P (16x64) @ G (64x128): 2 K-chunks x 8 column tiles = 16 WMMAs
        v16bf PA[2];
#pragma unroll
        for (int kb = 0; kb < 2; ++kb) {
            const __bf16* pr = pbase + l15 * 72 + kb * 32 + sel;
            PA[kb] = ld2(pr, pr + 16);
        }
#pragma unroll
        for (int ct = 0; ct < 8; ++ct) {
            const __bf16* gr = sG + (ct * 16 + l15) * 72 + (hi ? 8 : 0);
#pragma unroll
            for (int kb = 0; kb < 2; ++kb) {
                v16bf Bg = ld2(gr + kb * 32, gr + kb * 32 + 16);
                acc[ct] = __builtin_amdgcn_wmma_f32_16x16x32_bf16(
                    false, PA[kb], false, Bg, (short)0, acc[ct], false, false);
            }
        }
    }

    // normalize and store O as bf16 [b][p][128]
    float inv[8];
#pragma unroll
    for (int v = 0; v < 8; ++v) inv[v] = 1.0f / lrow[v];
    const size_t obase = ((size_t)b * kHW + rowbase) * kCG;
#pragma unroll
    for (int ct = 0; ct < 8; ++ct)
#pragma unroll
        for (int v = 0; v < 8; ++v) {
            int row = v + (hi ? 8 : 0);
            OW[obase + (size_t)row * kCG + ct * 16 + l15] =
                (__bf16)(acc[ct][v] * inv[v]);
        }
}

// ---------------------------------------------------------------------------
// Kernel 3: final 1x1 conv (K=128 -> 256) + BN + SiLU + residual, fp32 out.
// Block: 256 threads = 64 pixels x 4 groups of 64 output channels.
// ---------------------------------------------------------------------------
__global__ __launch_bounds__(256) void outproj_kernel(
    const __bf16* __restrict__ O, const float* __restrict__ w_out,
    const float* __restrict__ g_out, const float* __restrict__ b_out,
    const float* __restrict__ x, float* __restrict__ out)
{
    __shared__ __align__(16) __bf16 ldsO[64][40];
    const int blk = blockIdx.x;
    const int b = blk / kNT, pt = blk % kNT;
    const int p0 = pt * kTP;
    const int t = threadIdx.x;
    const int pix = t & 63, grp = t >> 6;

    float acc[64];
#pragma unroll
    for (int j = 0; j < 64; ++j) acc[j] = 0.0f;

    for (int kk = 0; kk < kCG; kk += 32) {
        __syncthreads();
        {
            int r = t >> 2, ch = t & 3;   // 64 rows x 4 x 16B chunks
            copy16_g2l(&ldsO[r][ch * 8],
                       O + ((size_t)b * kHW + p0 + r) * kCG + kk + ch * 8);
        }
        async_copy_wait();
        __syncthreads();
#pragma unroll 4
        for (int r = 0; r < 32; ++r) {
            float ov = (float)ldsO[pix][r];
            int k = kk + r;
#pragma unroll
            for (int j = 0; j < 64; ++j)
                acc[j] = fmaf(ov, w_out[(grp * 64 + j) * kCG + k], acc[j]);
        }
    }

#pragma unroll
    for (int j = 0; j < 64; ++j) {
        int o = grp * 64 + j;
        float y = silu(acc[j] * g_out[o] * kRS + b_out[o]);
        size_t idx = ((size_t)(b * kC1 + o)) * kHW + p0 + pix;
        out[idx] = y + x[idx];
    }
}

// ---------------------------------------------------------------------------
extern "C" void kernel_launch(void* const* d_in, const int* in_sizes, int n_in,
                              void* d_out, int out_size, void* d_ws, size_t ws_size,
                              hipStream_t stream) {
    const float* x     = (const float*)d_in[0];
    const float* w_th  = (const float*)d_in[1];
    const float* g_th  = (const float*)d_in[2];
    const float* b_th  = (const float*)d_in[3];
    const float* w_ph  = (const float*)d_in[4];
    const float* g_ph  = (const float*)d_in[5];
    const float* b_ph  = (const float*)d_in[6];
    const float* w_g   = (const float*)d_in[7];
    const float* g_g   = (const float*)d_in[8];
    const float* b_g   = (const float*)d_in[9];
    const float* w_out = (const float*)d_in[10];
    const float* g_out = (const float*)d_in[11];
    const float* b_out = (const float*)d_in[12];
    float* out = (float*)d_out;

    auto alignup = [](size_t v) { return (v + 255) & ~(size_t)255; };
    char* w = (char*)d_ws;
    __bf16* thW = (__bf16*)w; w += alignup((size_t)kB * kHW * kCI * 2);
    __bf16* phW = (__bf16*)w; w += alignup((size_t)kB * kHW * kCI * 2);
    __bf16* gW  = (__bf16*)w; w += alignup((size_t)kB * kCG * kHW * 2);
    __bf16* OW  = (__bf16*)w; w += alignup((size_t)kB * kHW * kCG * 2);

    dim3 grid(kB * kNT);   // 392 blocks
    proj_kernel<<<grid, 256, 0, stream>>>(x, w_th, w_ph, w_g,
                                          g_th, b_th, g_ph, b_ph, g_g, b_g,
                                          thW, phW, gW);
    attn_kernel<<<grid, 128, 0, stream>>>(thW, phW, gW, OW);
    outproj_kernel<<<grid, 256, 0, stream>>>(OW, w_out, g_out, b_out, x, out);
}
